// SEMMBlock_53815940219183
// MI455X (gfx1250) — matrix operand order repaired and checked by reading, not compiled, not run on hardware
//
#include <hip/hip_runtime.h>

// ---------------- problem constants (from reference) ----------------
#define DTOK   512
#define NEXP   8
#define FDIM   2048
#define NHEAD  4
#define DHEAD  128
#define BATCH  4
#define SEQ    1024
#define NTOK   (BATCH * SEQ)   // 4096

typedef __attribute__((ext_vector_type(16))) _Float16 v16h;
typedef __attribute__((ext_vector_type(8)))  _Float16 v8h;
typedef __attribute__((ext_vector_type(4)))  _Float16 v4h;
typedef __attribute__((ext_vector_type(8)))  float    v8f;

typedef unsigned int u32;
typedef unsigned long long u64;
typedef __attribute__((ext_vector_type(4))) u32 u32x4;
typedef __attribute__((ext_vector_type(4))) int i32x4;
typedef __attribute__((ext_vector_type(8))) int i32x8;

union FragA { v16h v; v8h h[2]; };

#if __has_builtin(__builtin_amdgcn_tensor_load_to_lds) && \
    __has_builtin(__builtin_amdgcn_s_wait_tensorcnt)
#define HAS_TDM 1
#else
#define HAS_TDM 0
#endif

__device__ __forceinline__ v8f wmma16x16x32(v16h a, v16h b, v8f c) {
  // 8 args: (neg_a, A, neg_b, B, c_mod, C, reuse_a, reuse_b)
  return __builtin_amdgcn_wmma_f32_16x16x32_f16(false, a, false, b, (short)0, c,
                                                false, false);
}

#if HAS_TDM
// ---------------------------------------------------------------------
// Tensor Data Mover: load one [128 rows x 32 cols] f16 tile (row stride =
// rowStride elements) from global into LDS at ldsAddr. D# per ISA ch.8.
// ---------------------------------------------------------------------
__device__ __forceinline__ void tdm_load_tile_f16(u32 ldsAddr,
                                                  const _Float16* gptr,
                                                  u32 rowStride) {
  const u64 ga = (u64)(uintptr_t)gptr;
  u32x4 g0;
  g0.x = 1u;                                            // count=1 (valid D#)
  g0.y = ldsAddr;                                       // LDS byte address
  g0.z = (u32)ga;                                       // global_addr[31:0]
  g0.w = (u32)((ga >> 32) & 0x1FFFFFFu) | (2u << 30);   // addr[56:32]|type=2
  i32x8 g1;
  g1[0] = (int)(1u << 16);                              // data_size=1 -> 2B
  g1[1] = (int)((rowStride & 0xFFFFu) << 16);           // tensor_dim0[15:0]
  g1[2] = (int)(((rowStride >> 16) & 0xFFFFu) | (128u << 16)); // dim0hi|dim1lo
  g1[3] = (int)(32u << 16);                             // dim1 hi | tile_dim0=32
  g1[4] = (int)128u;                                    // tile_dim1=128
  g1[5] = (int)rowStride;                               // dim0_stride[31:0]
  g1[6] = 0;
  g1[7] = 0;
  const i32x4 z4 = {0, 0, 0, 0};
#if defined(__clang_major__) && (__clang_major__ >= 23)
  const i32x8 z8 = {0, 0, 0, 0, 0, 0, 0, 0};
  __builtin_amdgcn_tensor_load_to_lds(g0, g1, z4, z4, z8, 0);
#else
  __builtin_amdgcn_tensor_load_to_lds(g0, g1, z4, z4, 0);
#endif
}
#endif  // HAS_TDM

// =====================================================================
// f32 -> f16 cast (vectorized), feeds the TDM-staged GEMM A operands
// =====================================================================
__global__ __launch_bounds__(256) void cast_f16_kernel(
    const float* __restrict__ in, _Float16* __restrict__ out, int n4) {
  const int i = blockIdx.x * 256 + threadIdx.x;
  if (i >= n4) return;
  const float4 f = ((const float4*)in)[i];
  v4h h;
  h.x = (_Float16)f.x; h.y = (_Float16)f.y;
  h.z = (_Float16)f.z; h.w = (_Float16)f.w;
  ((v4h*)out)[i] = h;
}

// =====================================================================
// Router: logits = x @ gate_w + gate_b; top-2 mask; relu; scale-and-fire
// =====================================================================
__global__ __launch_bounds__(256) void router_kernel(
    const float* __restrict__ x, const float* __restrict__ gw,
    const float* __restrict__ gb, float* __restrict__ logits,
    float* __restrict__ rw) {
  const int t = blockIdx.x * blockDim.x + threadIdx.x;
  if (t >= NTOK) return;
  float acc[NEXP];
#pragma unroll
  for (int e = 0; e < NEXP; ++e) acc[e] = gb[e];
  const float* xp = x + (size_t)t * DTOK;
  for (int k = 0; k < DTOK; ++k) {
    const float xv = xp[k];
    const float* g = gw + (size_t)k * NEXP;
#pragma unroll
    for (int e = 0; e < NEXP; ++e) acc[e] = fmaf(xv, g[e], acc[e]);
  }
  int i1 = 0; float m1 = acc[0];
#pragma unroll
  for (int e = 1; e < NEXP; ++e) if (acc[e] > m1) { m1 = acc[e]; i1 = e; }
  int i2 = -1; float m2 = -__builtin_inff();
#pragma unroll
  for (int e = 0; e < NEXP; ++e)
    if (e != i1 && acc[e] > m2) { m2 = acc[e]; i2 = e; }
#pragma unroll
  for (int e = 0; e < NEXP; ++e) {
    logits[(size_t)t * NEXP + e] = acc[e];
    float mem = (e == i1 || e == i2) ? fmaxf(acc[e], 0.0f) : 0.0f;
    float out = 0.0f;
    float thr = 4.0f;
#pragma unroll
    for (int lvl = 0; lvl < 4; ++lvl) {
      if (mem >= thr) { out += thr; mem -= thr; }
      thr *= 0.5f;
    }
    rw[(size_t)t * NEXP + e] = out;
  }
}

// =====================================================================
// WMMA GEMM: C[M,N] = act(A[M,K](f16) @ W[K,N](f32) + bias[N])
//   block = 256 threads = 8 waves; workgroup tile 128x64; wave tile 32x32.
//   A tiles: Tensor Data Mover -> LDS, double buffered on TENSORcnt.
//   W tiles: register-staged + f32->f16 convert (single LDS buffer).
// =====================================================================
template <int ACT>
__global__ __launch_bounds__(256) void gemm_wmma_kernel(
    const _Float16* __restrict__ A, const float* __restrict__ W,
    const float* __restrict__ bias, float* __restrict__ C,
    int M, int K, int Nn) {
  __shared__ _Float16 As[2][128][32];  // packed [m][k] tile (TDM destination)
  __shared__ _Float16 Bs[64][40];      // transposed [n][k], padded stride

  const int tid  = threadIdx.x;
  const int wid  = tid >> 5;
  const int lane = tid & 31;
  const int l16  = lane & 15;
  const int hlf  = lane >> 4;
  const int wm = (wid >> 1) * 32;   // wave M offset (0..96)
  const int wn = (wid & 1) * 32;    // wave N offset (0,32)
  const int m0 = blockIdx.x * 128;
  const int n0 = blockIdx.y * 64;

  // W-tile cooperative stage mapping
  const int bRow = tid >> 3;          // 0..31 (k)
  const int bCol = (tid & 7) * 8;     // 0..56 (n)

  v8f acc[2][2] = {};
  float4 bR[2];

  const _Float16* aTile = A + (size_t)m0 * K;     // 128 rows, stride K
  const float* wBase = W + (size_t)bRow * Nn + n0 + bCol;

  bR[0] = *(const float4*)(wBase);
  bR[1] = *(const float4*)(wBase + 4);

#if HAS_TDM
  const u32 ldsA0 = (u32)(uintptr_t)&As[0][0][0];
  const u32 ldsA1 = (u32)(uintptr_t)&As[1][0][0];
  if (wid == 0) tdm_load_tile_f16(ldsA0, aTile, (u32)K);  // tile 0 in flight
#else
  const int aRow = tid >> 1;          // 0..127
  const int aC   = (tid & 1) * 16;    // 0 / 16
#endif

  const int KT = K / 32;
  for (int kt = 0; kt < KT; ++kt) {
    const int buf = kt & 1;
    // ---- W tile: registers -> LDS (convert to f16, transposed) ----
#pragma unroll
    for (int i = 0; i < 2; ++i) {
      const float4 f = bR[i];
      Bs[bCol + 4 * i + 0][bRow] = (_Float16)f.x;
      Bs[bCol + 4 * i + 1][bRow] = (_Float16)f.y;
      Bs[bCol + 4 * i + 2][bRow] = (_Float16)f.z;
      Bs[bCol + 4 * i + 3][bRow] = (_Float16)f.w;
    }
#if HAS_TDM
    // ---- A tile: TDM pipeline (issue next, wait for current) ----
    if (wid == 0) {
      if (kt + 1 < KT) {
        tdm_load_tile_f16(buf ? ldsA0 : ldsA1,
                          aTile + (size_t)(kt + 1) * 32, (u32)K);
        __builtin_amdgcn_s_wait_tensorcnt((short)1);
      } else {
        __builtin_amdgcn_s_wait_tensorcnt((short)0);
      }
    }
#else
    // ---- A tile fallback: direct f16 global -> LDS copy ----
    {
      const _Float16* ap = aTile + (size_t)aRow * K + (size_t)kt * 32 + aC;
      const v8h a0 = *(const v8h*)(ap);
      const v8h a1 = *(const v8h*)(ap + 8);
      *(v8h*)&As[buf][aRow][aC] = a0;
      *(v8h*)&As[buf][aRow][aC + 8] = a1;
    }
#endif
    __syncthreads();
    // prefetch next W tile into registers (overlaps WMMA below)
    if (kt + 1 < KT) {
      const float* wp = wBase + (size_t)(kt + 1) * 32 * Nn;
      bR[0] = *(const float4*)(wp);
      bR[1] = *(const float4*)(wp + 4);
      __builtin_prefetch(wp + (size_t)32 * Nn, 0, 1);  // global_prefetch_b8
    }
    // ---- fragments per ISA 7.12.2 ----
    v16h af[2]; v16h bf[2];
#pragma unroll
    for (int mt = 0; mt < 2; ++mt) {
      FragA fa;
      const int r = wm + mt * 16 + l16;
      fa.h[0] = *(const v8h*)&As[buf][r][hlf * 8];
      fa.h[1] = *(const v8h*)&As[buf][r][16 + hlf * 8];
      af[mt] = fa.v;
    }
#pragma unroll
    for (int nt = 0; nt < 2; ++nt) {
      FragA fb;
      const int c = wn + nt * 16 + l16;
      fb.h[0] = *(const v8h*)&Bs[c][hlf * 8];
      fb.h[1] = *(const v8h*)&Bs[c][16 + hlf * 8];
      bf[nt] = fb.v;
    }
#pragma unroll
    for (int mt = 0; mt < 2; ++mt)
#pragma unroll
      for (int nt = 0; nt < 2; ++nt)
        acc[mt][nt] = wmma16x16x32(af[mt], bf[nt], acc[mt][nt]);
    __syncthreads();
  }

  // epilogue: C-frag layout M = v + 8*hlf, N = l16
#pragma unroll
  for (int mt = 0; mt < 2; ++mt) {
#pragma unroll
    for (int nt = 0; nt < 2; ++nt) {
      const int col = n0 + wn + nt * 16 + l16;
      const float bvv = bias[col];
      const int rbase = m0 + wm + mt * 16 + 8 * hlf;
#pragma unroll
      for (int v = 0; v < 8; ++v) {
        float val = acc[mt][nt][v] + bvv;
        if (ACT == 1) val = fmaxf(val, 0.0f);
        C[(size_t)(rbase + v) * Nn + col] = val;
      }
    }
  }
}

// =====================================================================
// Flash attention: one block per (b, h, 128-query block); 8 waves,
// each wave owns 16 query rows. Online softmax; no score materialization.
// =====================================================================
__global__ __launch_bounds__(256) void attn_kernel(
    const float* __restrict__ Q, const float* __restrict__ Kc,
    const float* __restrict__ V, float* __restrict__ O) {
  __shared__ _Float16 Ks[32][136];     // [s][d]  (row-major K block)
  __shared__ _Float16 Vs[128][40];     // [d][s]  (transposed V block)
  __shared__ _Float16 Ps[8][16][40];   // per-wave P tile [q][s]

  const int tid  = threadIdx.x;
  const int wid  = tid >> 5;
  const int lane = tid & 31;
  const int l16  = lane & 15;
  const int hlf  = lane >> 4;
  const int bi = blockIdx.x;
  const int qb = bi & 7;
  const int h  = (bi >> 3) & 3;
  const int b  = bi >> 5;
  const int q0 = qb * 128;
  const int colBase = h * DHEAD;

  // Q fragments (16 rows x 128), pre-scaled by 1/sqrt(dh)
  const float qscale = 0.08838834764831845f;  // 1/sqrt(128)
  v16h aQ[4];
  {
    const int qtok = b * SEQ + q0 + wid * 16 + l16;
    const float* qp = Q + (size_t)qtok * DTOK + colBase;
#pragma unroll
    for (int t = 0; t < 4; ++t) {
      FragA fa;
#pragma unroll
      for (int c = 0; c < 2; ++c) {
        const int d0 = t * 32 + c * 16 + hlf * 8;
        const float4 f0 = *(const float4*)(qp + d0);
        const float4 f1 = *(const float4*)(qp + d0 + 4);
        v8h hh;
        hh[0] = (_Float16)(f0.x * qscale); hh[1] = (_Float16)(f0.y * qscale);
        hh[2] = (_Float16)(f0.z * qscale); hh[3] = (_Float16)(f0.w * qscale);
        hh[4] = (_Float16)(f1.x * qscale); hh[5] = (_Float16)(f1.y * qscale);
        hh[6] = (_Float16)(f1.z * qscale); hh[7] = (_Float16)(f1.w * qscale);
        fa.h[c] = hh;
      }
      aQ[t] = fa.v;
    }
  }

  v8f accO[8] = {};
  float mrow[8], lrow[8];
#pragma unroll
  for (int v = 0; v < 8; ++v) { mrow[v] = -__builtin_inff(); lrow[v] = 0.0f; }

  const int sRow  = tid >> 3;        // 0..31
  const int dBase = (tid & 7) * 16;  // 0..112

  for (int sb = 0; sb < SEQ / 32; ++sb) {
    const int s0 = sb * 32;
    __syncthreads();
    // stage K row-major and V transposed into LDS (f32 -> f16)
    {
      const int ktok = b * SEQ + s0 + sRow;
      const float* kp = Kc + (size_t)ktok * DTOK + colBase + dBase;
      const float* vp = V  + (size_t)ktok * DTOK + colBase + dBase;
#pragma unroll
      for (int i = 0; i < 4; ++i) {
        const float4 f = *(const float4*)(kp + 4 * i);
        v4h hk;
        hk.x = (_Float16)f.x; hk.y = (_Float16)f.y;
        hk.z = (_Float16)f.z; hk.w = (_Float16)f.w;
        *(v4h*)&Ks[sRow][dBase + 4 * i] = hk;
        const float4 g = *(const float4*)(vp + 4 * i);
        Vs[dBase + 4 * i + 0][sRow] = (_Float16)g.x;
        Vs[dBase + 4 * i + 1][sRow] = (_Float16)g.y;
        Vs[dBase + 4 * i + 2][sRow] = (_Float16)g.z;
        Vs[dBase + 4 * i + 3][sRow] = (_Float16)g.w;
      }
    }
    __syncthreads();

    // scores[16x32] = Q(16x128) @ K^T(128x32) : 8 WMMA over 4 k-steps
    v8f c0 = {}, c1 = {};
#pragma unroll
    for (int t = 0; t < 4; ++t) {
      FragA b0, b1;
      const int kd = t * 32 + hlf * 8;
      b0.h[0] = *(const v8h*)&Ks[l16][kd];
      b0.h[1] = *(const v8h*)&Ks[l16][kd + 16];
      b1.h[0] = *(const v8h*)&Ks[16 + l16][kd];
      b1.h[1] = *(const v8h*)&Ks[16 + l16][kd + 16];
      c0 = wmma16x16x32(aQ[t], b0.v, c0);
      c1 = wmma16x16x32(aQ[t], b1.v, c1);
    }

    // online softmax: rows live as (v, lane-half) slots; 16-lane butterflies
    float al[8];
#pragma unroll
    for (int v = 0; v < 8; ++v) {
      float rm = fmaxf(c0[v], c1[v]);
#pragma unroll
      for (int msk = 1; msk < 16; msk <<= 1)
        rm = fmaxf(rm, __shfl_xor(rm, msk, 32));
      const float mn = fmaxf(mrow[v], rm);
      al[v]   = __expf(mrow[v] - mn);
      mrow[v] = mn;
      c0[v] = __expf(c0[v] - mn);
      c1[v] = __expf(c1[v] - mn);
      float rs = c0[v] + c1[v];
#pragma unroll
      for (int msk = 1; msk < 16; msk <<= 1) rs += __shfl_xor(rs, msk, 32);
      lrow[v] = lrow[v] * al[v] + rs;
    }
#pragma unroll
    for (int dt = 0; dt < 8; ++dt)
#pragma unroll
      for (int v = 0; v < 8; ++v) accO[dt][v] *= al[v];

    // P: C-fragment -> LDS -> A-fragment re-layout (same wave only)
#pragma unroll
    for (int v = 0; v < 8; ++v) {
      Ps[wid][v + 8 * hlf][l16]      = (_Float16)c0[v];
      Ps[wid][v + 8 * hlf][16 + l16] = (_Float16)c1[v];
    }
    asm volatile("s_wait_dscnt 0" ::: "memory");  // CDNA5 split DS counter wait
    FragA ap;
    ap.h[0] = *(const v8h*)&Ps[wid][l16][hlf * 8];
    ap.h[1] = *(const v8h*)&Ps[wid][l16][16 + hlf * 8];

    // O(16x128) += P(16x32) @ V(32x128) : 8 WMMA
#pragma unroll
    for (int dt = 0; dt < 8; ++dt) {
      FragA bv;
      const int r = dt * 16 + l16;
      bv.h[0] = *(const v8h*)&Vs[r][hlf * 8];
      bv.h[1] = *(const v8h*)&Vs[r][16 + hlf * 8];
      accO[dt] = wmma16x16x32(ap.v, bv.v, accO[dt]);
    }
  }

  // normalize by softmax denominator and store
#pragma unroll
  for (int v = 0; v < 8; ++v) {
    const float inv = 1.0f / lrow[v];
    const int tok = b * SEQ + q0 + wid * 16 + 8 * hlf + v;
    float* op = O + (size_t)tok * DTOK + colBase + l16;
#pragma unroll
    for (int dt = 0; dt < 8; ++dt) op[dt * 16] = accO[dt][v] * inv;
  }
}

// =====================================================================
// h = LayerNorm(x + o) * s + b    (one block per token row of 512)
// =====================================================================
__global__ __launch_bounds__(256) void add_ln_kernel(
    const float* __restrict__ x, const float* __restrict__ o,
    const float* __restrict__ sc, const float* __restrict__ bi,
    float* __restrict__ out) {
  __shared__ float red[256];
  const int t = blockIdx.x;
  const int c = threadIdx.x;
  const size_t base = (size_t)t * DTOK;
  const float a0 = x[base + c] + o[base + c];
  const float a1 = x[base + c + 256] + o[base + c + 256];
  red[c] = a0 + a1;
  __syncthreads();
  for (int st = 128; st > 0; st >>= 1) {
    if (c < st) red[c] += red[c + st];
    __syncthreads();
  }
  const float mu = red[0] * (1.0f / DTOK);
  __syncthreads();
  const float d0 = a0 - mu, d1 = a1 - mu;
  red[c] = d0 * d0 + d1 * d1;
  __syncthreads();
  for (int st = 128; st > 0; st >>= 1) {
    if (c < st) red[c] += red[c + st];
    __syncthreads();
  }
  const float r = rsqrtf(red[0] * (1.0f / DTOK) + 1e-5f);
  out[base + c]       = d0 * r * sc[c] + bi[c];
  out[base + c + 256] = d1 * r * sc[c + 256] + bi[c + 256];
}

// =====================================================================
// final += rw[:,e] * LayerNorm(h + f)   (init writes, others accumulate)
// =====================================================================
__global__ __launch_bounds__(256) void final_ln_accum_kernel(
    const float* __restrict__ hb, const float* __restrict__ f,
    const float* __restrict__ sc, const float* __restrict__ bi,
    const float* __restrict__ rw, int e, float* __restrict__ out, int init) {
  __shared__ float red[256];
  const int t = blockIdx.x;
  const int c = threadIdx.x;
  const size_t base = (size_t)t * DTOK;
  const float a0 = hb[base + c] + f[base + c];
  const float a1 = hb[base + c + 256] + f[base + c + 256];
  red[c] = a0 + a1;
  __syncthreads();
  for (int st = 128; st > 0; st >>= 1) {
    if (c < st) red[c] += red[c + st];
    __syncthreads();
  }
  const float mu = red[0] * (1.0f / DTOK);
  __syncthreads();
  const float d0 = a0 - mu, d1 = a1 - mu;
  red[c] = d0 * d0 + d1 * d1;
  __syncthreads();
  for (int st = 128; st > 0; st >>= 1) {
    if (c < st) red[c] += red[c + st];
    __syncthreads();
  }
  const float r = rsqrtf(red[0] * (1.0f / DTOK) + 1e-5f);
  const float w = rw[(size_t)t * NEXP + e];
  const float y0 = (d0 * r * sc[c] + bi[c]) * w;
  const float y1 = (d1 * r * sc[c + 256] + bi[c + 256]) * w;
  if (init) {
    out[base + c] = y0;
    out[base + c + 256] = y1;
  } else {
    out[base + c] += y0;
    out[base + c + 256] += y1;
  }
}

// =====================================================================
extern "C" void kernel_launch(void* const* d_in, const int* in_sizes, int n_in,
                              void* d_out, int out_size, void* d_ws,
                              size_t ws_size, hipStream_t stream) {
  const float* x      = (const float*)d_in[0];
  const float* gate_w = (const float*)d_in[1];
  const float* gate_b = (const float*)d_in[2];
  const float* ln1_s  = (const float*)d_in[3];
  const float* ln1_b  = (const float*)d_in[4];
  const float* ln2_s  = (const float*)d_in[5];
  const float* ln2_b  = (const float*)d_in[6];
  const float* wq = (const float*)d_in[7];
  const float* wk = (const float*)d_in[8];
  const float* wv = (const float*)d_in[9];
  const float* wo = (const float*)d_in[10];
  const float* bq = (const float*)d_in[11];
  const float* bk = (const float*)d_in[12];
  const float* bv = (const float*)d_in[13];
  const float* bo = (const float*)d_in[14];
  const float* w1 = (const float*)d_in[15];
  const float* b1 = (const float*)d_in[16];
  const float* w2 = (const float*)d_in[17];
  const float* b2 = (const float*)d_in[18];

  float* outF = (float*)d_out;                 // final [4,1024,512]
  float* outL = outF + (size_t)NTOK * DTOK;    // logits [4,1024,8]

  float* ws = (float*)d_ws;
  size_t off = 0;
  float* rw  = ws + off; off += (size_t)NTOK * NEXP;
  float* Qb  = ws + off; off += (size_t)NTOK * DTOK;
  float* Kb  = ws + off; off += (size_t)NTOK * DTOK;
  float* Vb  = ws + off; off += (size_t)NTOK * DTOK;
  float* AOb = ws + off; off += (size_t)NTOK * DTOK;
  float* Hb  = ws + off; off += (size_t)NTOK * DTOK;
  float* F1b = ws + off; off += (size_t)NTOK * FDIM;
  // f16 shadows of GEMM A-operands (TDM copies raw bytes, so pre-cast)
  _Float16* xh  = (_Float16*)(ws + off); off += (size_t)NTOK * DTOK / 2;
  _Float16* AOh = (_Float16*)(ws + off); off += (size_t)NTOK * DTOK / 2;
  _Float16* Hh  = (_Float16*)(ws + off); off += (size_t)NTOK * DTOK / 2;
  _Float16* F1h = (_Float16*)(ws + off); off += (size_t)NTOK * FDIM / 2;
  float* O2b = Qb;  // reuse (Q dead after attention)
  float* F2b = Kb;  // reuse (K dead after attention)

  router_kernel<<<NTOK / 256, 256, 0, stream>>>(x, gate_w, gate_b, outL, rw);

  const dim3 blk(256);
  const dim3 gD(NTOK / 128, DTOK / 64);   // (32, 8)
  const dim3 gF(NTOK / 128, FDIM / 64);   // (32, 32)
  const int nD4 = NTOK * DTOK / 4;        // cast quads, D-sized buffers
  const int nF4 = NTOK * FDIM / 4;        // cast quads, F-sized buffer

  cast_f16_kernel<<<nD4 / 256, blk, 0, stream>>>(x, xh, nD4);

  for (int e = 0; e < NEXP; ++e) {
    const size_t wOff  = (size_t)e * DTOK * DTOK;
    const size_t dOff  = (size_t)e * DTOK;
    const size_t w1Off = (size_t)e * DTOK * FDIM;
    const size_t fOff  = (size_t)e * FDIM;
    gemm_wmma_kernel<0><<<gD, blk, 0, stream>>>(xh, wq + wOff, bq + dOff, Qb,
                                                NTOK, DTOK, DTOK);
    gemm_wmma_kernel<0><<<gD, blk, 0, stream>>>(xh, wk + wOff, bk + dOff, Kb,
                                                NTOK, DTOK, DTOK);
    gemm_wmma_kernel<0><<<gD, blk, 0, stream>>>(xh, wv + wOff, bv + dOff, Vb,
                                                NTOK, DTOK, DTOK);
    attn_kernel<<<BATCH * NHEAD * (SEQ / 128), blk, 0, stream>>>(Qb, Kb, Vb,
                                                                 AOb);
    cast_f16_kernel<<<nD4 / 256, blk, 0, stream>>>(AOb, AOh, nD4);
    gemm_wmma_kernel<0><<<gD, blk, 0, stream>>>(AOh, wo + wOff, bo + dOff, O2b,
                                                NTOK, DTOK, DTOK);
    add_ln_kernel<<<NTOK, blk, 0, stream>>>(x, O2b, ln1_s + dOff, ln1_b + dOff,
                                            Hb);
    cast_f16_kernel<<<nD4 / 256, blk, 0, stream>>>(Hb, Hh, nD4);
    gemm_wmma_kernel<1><<<gF, blk, 0, stream>>>(Hh, w1 + w1Off, b1 + fOff, F1b,
                                                NTOK, DTOK, FDIM);
    cast_f16_kernel<<<nF4 / 256, blk, 0, stream>>>(F1b, F1h, nF4);
    gemm_wmma_kernel<0><<<gD, blk, 0, stream>>>(F1h, w2 + w1Off, b2 + dOff,
                                                F2b, NTOK, FDIM, DTOK);
    final_ln_accum_kernel<<<NTOK, blk, 0, stream>>>(
        Hb, F2b, ln2_s + dOff, ln2_b + dOff, rw, e, outF, e == 0 ? 1 : 0);
  }
  (void)in_sizes; (void)n_in; (void)out_size; (void)ws_size;
}
